// LSTM_24876450578598
// MI455X (gfx1250) — matrix operand order repaired and checked by reading, not compiled, not run on hardware
//
#include <hip/hip_runtime.h>
#include <hip/hip_bf16.h>
#include <cstdint>
#include <cstddef>

// ---------------------------------------------------------------------------
// Bi-LSTM tagger, batch=1, T=4096, EMB=512, HID=1024, NTAGS=5.
// Forward dir: full serial recurrence. Persistent WMMA kernel; the 12MB
//   gate-weight matrix lives in VGPRs (48 x v16bf per lane, VGPR-MSB range)
//   for all 4096 steps; A-fragment loads are software-pipelined (2-deep
//   rotation) so L0 latency hides behind WMMA execution; one device-wide
//   barrier per step.
// Backward dir: reference only consumes hb[-1] == one cell step on x[T-1].
// ---------------------------------------------------------------------------

typedef __attribute__((ext_vector_type(16))) __bf16 v16bf;
typedef __attribute__((ext_vector_type(8)))  __bf16 v8bf;
typedef __attribute__((ext_vector_type(8)))  float  v8f;

#define T_SEQ  4096
#define EMB    512
#define HID    1024
#define G4     4096        // 4*HID gate rows
#define KTOT   1536        // EMB + HID (fused GEMV)
#define KTILES (KTOT / 32) // 48 WMMA K-tiles
#define NBLK   32          // persistent grid
#define NTHR   256         // 8 waves/block -> 256 waves -> 256 tiles of 16 cols
#define NTAGS  5

// -------------------- device-wide barrier (persistent kernel) --------------
__device__ __forceinline__ void gbar(unsigned* bar, unsigned nblocks) {
    __threadfence();            // release (global_wb)
    __syncthreads();
    if (threadIdx.x == 0) {
        volatile unsigned* cnt = bar;
        volatile unsigned* gen = bar + 1;
        unsigned g = *gen;
        if (atomicAdd(bar, 1u) == nblocks - 1u) {
            *cnt = 0u;
            __threadfence();
            *gen = g + 1u;
        } else {
            while (*gen == g) { __builtin_amdgcn_s_sleep(2); }
        }
    }
    __syncthreads();
    __threadfence();            // acquire (global_inv: drop stale L0 lines)
}

__device__ __forceinline__ float sigf(float x) { return 1.0f / (1.0f + expf(-x)); }

// -------------------- prep: weights->bf16 (gate-interleaved), embed gather --
__global__ void prep_kernel(const int* __restrict__ words,
                            const float* __restrict__ emb,
                            const float* __restrict__ Wih,
                            const float* __restrict__ Whh,
                            const float* __restrict__ bih,
                            const float* __restrict__ bhh,
                            __bf16* __restrict__ Wcat,   // [4096][1536], row r=4u+g
                            __bf16* __restrict__ xb,     // [4096][512]
                            float*  __restrict__ bias,   // [4096] permuted
                            unsigned* __restrict__ bar) {
    const int b = blockIdx.x;
    const int t = threadIdx.x;
    if (b < G4) {
        // permuted row r = 4*u + g  <- source row  g*HID + u  (gate order i,f,g,o)
        const int g = b & 3, u = b >> 2;
        const int src = g * HID + u;
        const float* wi = Wih + (size_t)src * EMB;
        const float* wh = Whh + (size_t)src * HID;
        __bf16* dst = Wcat + (size_t)b * KTOT;
        for (int k = t; k < EMB; k += NTHR) dst[k]       = (__bf16)wi[k];
        for (int k = t; k < HID; k += NTHR) dst[EMB + k] = (__bf16)wh[k];
        if (t == 0) bias[b] = bih[src] + bhh[src];
    } else if (b < 2 * G4) {
        const int tt = b - G4;
        const float* e = emb + (size_t)words[tt] * EMB;
        __bf16* dst = xb + (size_t)tt * EMB;
        for (int k = t; k < EMB; k += NTHR) dst[k] = (__bf16)e[k];
    } else {
        if (t < 2) bar[t] = 0u;
    }
}

// -------------------- forward recurrence: persistent WMMA GEMV -------------
__global__ __launch_bounds__(NTHR, 1)
void lstm_fwd_kernel(const __bf16* __restrict__ Wcat,
                     const __bf16* __restrict__ xb,
                     const float*  __restrict__ bias,
                     __bf16* __restrict__ vbuf,   // ping-pong [2][1536] = [x_t ; h]
                     float*  __restrict__ hfv,    // [1024] final forward h
                     unsigned* __restrict__ bar) {
    const int tid  = threadIdx.x;
    const int lane = tid & 31;
    const int wave = ((int)blockIdx.x * NTHR + tid) >> 5;   // 0..255
    const int j0   = wave * 16;                             // gate-tile base

    // init v[0] = [x_0 ; 0]
    if (blockIdx.x == 0) {
        for (int k = tid; k < EMB; k += NTHR)        vbuf[k] = xb[k];
        for (int k = EMB + tid; k < KTOT; k += NTHR) vbuf[k] = (__bf16)0.0f;
    }

    // B fragment addressing (32x16 bf16 layout): lane holds column N=lane&15 =
    // Wcat row j0+N; lanes>=16 take K+16..K+31 of the 32-wide K tile.
    const __bf16* wrow = Wcat + (size_t)(j0 + (lane & 15)) * KTOT + ((lane & 16) ? 16 : 0);
    // A fragment (16x32 bf16, rows replicated = GEMV broadcast):
    // lane<16 -> K {0..15}; lane>=16 -> K {16..31}
    const int aoff = (lane & 16) ? 16 : 0;
    const float bj = bias[j0 + (lane & 15)];

    // ---- preload the whole B operand into registers (held 4096 steps) ----
    v16bf breg[KTILES];
    #pragma unroll
    for (int kt = 0; kt < KTILES; ++kt)
        breg[kt] = *(const v16bf*)(wrow + kt * 32);

    gbar(bar, NBLK);

    float creg = 0.0f;   // cell state lives in a register (lane owns its unit)

    for (int t = 0; t < T_SEQ; ++t) {
        const __bf16* vc = vbuf + (size_t)(t & 1) * KTOT + aoff;
        __bf16*       vn = vbuf + (size_t)((t + 1) & 1) * KTOT;

        // GEMV via WMMA. Two interleaved accumulators break the WMMA RAW
        // chain; a 2-deep rotating A buffer issues tile k+1's loads before
        // tile k's WMMA so L0 latency hides behind matrix execution.
        v8f acc0 = {}, acc1 = {};
        v8bf loA = *(const v8bf*)(vc);
        v8bf hiA = *(const v8bf*)(vc + 8);
        v8bf loB, hiB;
        #pragma unroll
        for (int kt = 0; kt < KTILES; kt += 2) {
            // prefetch tile kt+1 into buffer B, then consume buffer A
            loB = *(const v8bf*)(vc + (kt + 1) * 32);
            hiB = *(const v8bf*)(vc + (kt + 1) * 32 + 8);
            v16bf aA = __builtin_shufflevector(loA, hiA,
                         0,1,2,3,4,5,6,7,8,9,10,11,12,13,14,15);
            acc0 = __builtin_amdgcn_wmma_f32_16x16x32_bf16(
                       false, aA, false, breg[kt], (short)0, acc0, false, false);

            // prefetch tile kt+2 into buffer A, then consume buffer B
            if (kt + 2 < KTILES) {
                loA = *(const v8bf*)(vc + (kt + 2) * 32);
                hiA = *(const v8bf*)(vc + (kt + 2) * 32 + 8);
            }
            v16bf aB = __builtin_shufflevector(loB, hiB,
                         0,1,2,3,4,5,6,7,8,9,10,11,12,13,14,15);
            acc1 = __builtin_amdgcn_wmma_f32_16x16x32_bf16(
                       false, aB, false, breg[kt + 1], (short)0, acc1, false, false);
        }
        // D row 0 (all rows equal): lane n<16 holds gate value for column j0+n
        const float gv = acc0[0] + acc1[0] + bj;

        // gates interleaved per unit: quad {i,f,g,o}. Gather within the quad.
        const int qb = lane & ~3;
        const float iv = __shfl(gv, qb,     32);
        const float fv = __shfl(gv, qb | 1, 32);
        const float gg = __shfl(gv, qb | 2, 32);
        const float ov = __shfl(gv, qb | 3, 32);

        if (lane < 16 && (lane & 3) == 0) {
            const int u = (j0 + lane) >> 2;          // hidden unit owned by this lane
            creg = sigf(fv) * creg + sigf(iv) * tanhf(gg);
            const float h = sigf(ov) * tanhf(creg);
            vn[EMB + u] = (__bf16)h;                 // publish h for next step
            hfv[u] = h;                              // last write = hf[T-1]
        }
        // stage x_{t+1} into next v buffer (2 values per wave)
        if (lane < 2 && (t + 1) < T_SEQ) {
            const int k = wave * 2 + lane;
            vn[k] = xb[(size_t)(t + 1) * EMB + k];
        }
        gbar(bar, NBLK);
    }
}

// -------------------- backward: single cell on x[T-1] ----------------------
__global__ void bwd_gates_kernel(const int* __restrict__ words,
                                 const float* __restrict__ emb,
                                 const float* __restrict__ Wih_b,
                                 const float* __restrict__ bih_b,
                                 const float* __restrict__ bhh_b,
                                 float* __restrict__ gb) {
    const int j = blockIdx.x;            // 0..4095 (unpermuted source order)
    const int lane = threadIdx.x;        // one wave32
    const float* e = emb + (size_t)words[T_SEQ - 1] * EMB;
    const float* w = Wih_b + (size_t)j * EMB;
    float s = 0.0f;
    for (int k = lane; k < EMB; k += 32) s = fmaf(w[k], e[k], s);
    for (int o = 16; o > 0; o >>= 1) s += __shfl_down(s, o, 32);
    if (lane == 0) gb[j] = s + bih_b[j] + bhh_b[j];
}

__global__ void bwd_combine_kernel(const float* __restrict__ gb,
                                   float* __restrict__ hb) {
    const int u = blockIdx.x * blockDim.x + threadIdx.x;
    if (u < HID) {
        // c0 = 0  ->  c = sig(i)*tanh(g);  h = sig(o)*tanh(c)
        const float c = sigf(gb[u]) * tanhf(gb[2 * HID + u]);
        hb[u] = sigf(gb[3 * HID + u]) * tanhf(c);
    }
}

// -------------------- projection [2H] -> [NTAGS] ---------------------------
__global__ void proj_kernel(const float* __restrict__ hfv,
                            const float* __restrict__ hb,
                            const float* __restrict__ Wp,
                            const float* __restrict__ bp,
                            float* __restrict__ out) {
    __shared__ float red[NTHR];
    for (int n = 0; n < NTAGS; ++n) {
        float s = 0.0f;
        for (int k = threadIdx.x; k < 2 * HID; k += NTHR) {
            const float v = (k < HID) ? hfv[k] : hb[k - HID];
            s = fmaf(Wp[(size_t)n * 2 * HID + k], v, s);
        }
        red[threadIdx.x] = s;
        __syncthreads();
        for (int o = NTHR / 2; o > 0; o >>= 1) {
            if (threadIdx.x < o) red[threadIdx.x] += red[threadIdx.x + o];
            __syncthreads();
        }
        if (threadIdx.x == 0) out[n] = red[0] + bp[n];
        __syncthreads();
    }
}

// ---------------------------------------------------------------------------
extern "C" void kernel_launch(void* const* d_in, const int* in_sizes, int n_in,
                              void* d_out, int out_size, void* d_ws, size_t ws_size,
                              hipStream_t stream) {
    (void)in_sizes; (void)n_in; (void)out_size; (void)ws_size;

    const int*   words = (const int*)  d_in[0];
    const float* emb   = (const float*)d_in[1];
    const float* Wih_f = (const float*)d_in[2];
    const float* Whh_f = (const float*)d_in[3];
    const float* bih_f = (const float*)d_in[4];
    const float* bhh_f = (const float*)d_in[5];
    const float* Wih_b = (const float*)d_in[6];
    // d_in[7] (Whh_b) unused: backward h0 = 0, its Whh contribution is zero.
    const float* bih_b = (const float*)d_in[8];
    const float* bhh_b = (const float*)d_in[9];
    const float* Wp    = (const float*)d_in[10];
    const float* bp    = (const float*)d_in[11];
    float* out = (float*)d_out;

    // ---- workspace layout (~17.1 MB) ----
    char* base = (char*)d_ws;
    __bf16* Wcat = (__bf16*)base;                          // 4096*1536*2 = 12,582,912
    __bf16* xb   = Wcat + (size_t)G4 * KTOT;               //  4096*512*2 =  4,194,304
    __bf16* vbuf = xb   + (size_t)T_SEQ * EMB;             //   2*1536*2  =      6,144
    float*  bias = (float*)(base + 17u * 1024u * 1024u);   // f32 region @ 17 MiB
    float*  gb   = bias + G4;                              // 4096 f32
    float*  hfv  = gb   + G4;                              // 1024 f32
    float*  hb   = hfv  + HID;                             // 1024 f32
    unsigned* bar = (unsigned*)(hb + HID);                 // 2 u32

    prep_kernel<<<2 * G4 + 1, NTHR, 0, stream>>>(
        words, emb, Wih_f, Whh_f, bih_f, bhh_f, Wcat, xb, bias, bar);

    lstm_fwd_kernel<<<NBLK, NTHR, 0, stream>>>(Wcat, xb, bias, vbuf, hfv, bar);

    bwd_gates_kernel<<<G4, 32, 0, stream>>>(words, emb, Wih_b, bih_b, bhh_b, gb);
    bwd_combine_kernel<<<HID / NTHR, NTHR, 0, stream>>>(gb, hb);
    proj_kernel<<<1, NTHR, 0, stream>>>(hfv, hb, Wp, bp, out);
}
